// KnowledgeGraph_88192858456649
// MI455X (gfx1250) — matrix operand order repaired and checked by reading, not compiled, not run on hardware
//
#include <hip/hip_runtime.h>

#define B_      16
#define N_      1024
#define M_      4
#define D_      128
#define VOCAB_  32000
#define HOPS_   3
#define HEADS_  4
#define ALPHA_  0.2f
#define NEG_INF_ (-9e15f)

typedef __attribute__((ext_vector_type(16))) __bf16 v16bf;
typedef __attribute__((ext_vector_type(8)))  float  v8f;

union BF16x16 { v16bf v; unsigned short s[16]; uint4 u[2]; };
union US8     { unsigned short s[8]; uint4 u; };

// fp32 -> bf16 (round to nearest even), branch-free
__device__ __forceinline__ unsigned short f2bf(float x) {
    union { float f; unsigned u; } v; v.f = x;
    unsigned r = v.u + 0x7fffu + ((v.u >> 16) & 1u);
    return (unsigned short)(r >> 16);
}

__device__ __forceinline__ float lrelu(float x) {
    return x >= 0.f ? x : ALPHA_ * x;
}

// ---------------------------------------------------------------------------
// X[b,n,:] = sum_m emb[hop][story[b,n,m]] (+ dh_outputs scatter), stored bf16
// grid: B*N blocks, 128 threads
// ---------------------------------------------------------------------------
__global__ __launch_bounds__(128) void embed_kernel(
    const int* __restrict__ story, const int* __restrict__ kb,
    const int* __restrict__ cv, const float* __restrict__ dh,
    const float* __restrict__ emb, int hop,
    unsigned short* __restrict__ Xbf)
{
    int bn = blockIdx.x, d = threadIdx.x;
    int b = bn >> 10, n = bn & 1023;
    const int* st = story + bn * M_;
    float s = 0.f;
#pragma unroll
    for (int m = 0; m < M_; ++m) {
        int tok = st[m];
        s += emb[((size_t)hop * VOCAB_ + tok) * D_ + d];
    }
    int rel = n - (kb[b] - 1);
    if (rel >= 0 && rel < cv[b]) s += dh[((size_t)b * N_ + rel) * D_ + d];
    Xbf[(size_t)bn * D_ + d] = f2bf(s);
}

// ---------------------------------------------------------------------------
// Wh = X * W[hop,h]; output stored TRANSPOSED bf16: WhT[h,b,e,n].
// Also computes f1/f2 = Wh . a exactly from the f32 WMMA accumulators.
// grid: dim3(1024, HEADS), 256 threads (8 waves, one 16x16 tile each)
// ---------------------------------------------------------------------------
__global__ __launch_bounds__(256) void wh_gemm_kernel(
    const unsigned short* __restrict__ Xbf,
    const float* __restrict__ gatW,   // [HOPS+1, HEADS, D, D]
    const float* __restrict__ gata,   // [HOPS+1, HEADS, 2D]
    int hop,
    unsigned short* __restrict__ WhT, // [HEADS, B, D, N] bf16
    float* __restrict__ f1, float* __restrict__ f2)
{
    __shared__ unsigned short Wt[D_ * D_];     // W transposed: Wt[e*128 + d]
    __shared__ float pf1[8][16], pf2[8][16];

    const int h = blockIdx.y;
    const int rtile = blockIdx.x;
    const int tid = threadIdx.x;

    const float* Wsrc = gatW + (size_t)(hop * HEADS_ + h) * D_ * D_;
    for (int idx = tid; idx < D_ * D_; idx += 256) {
        int d = idx >> 7, e = idx & 127;
        Wt[e * D_ + d] = f2bf(Wsrc[idx]);
    }
    __syncthreads();

    const int wave = tid >> 5, lane = tid & 31;
    const int half = lane >> 4, lr = lane & 15;
    const int c0 = wave * 16;          // output column tile base (e)
    const int r0 = rtile * 16;         // global row base over B*N
    const int b  = r0 >> 10;
    const int n0 = r0 & 1023;

    v8f acc = {};
    const unsigned short* arow = Xbf + (size_t)(r0 + lr) * D_;
#pragma unroll
    for (int kk = 0; kk < D_; kk += 32) {
        BF16x16 af, bf;
        af.u[0] = *(const uint4*)(arow + kk + half * 8);
        af.u[1] = *(const uint4*)(arow + kk + half * 8 + 16);
        const unsigned short* bp = &Wt[(c0 + lr) * D_ + kk + half * 8];
        bf.u[0] = *(const uint4*)(bp);
        bf.u[1] = *(const uint4*)(bp + 16);
        acc = __builtin_amdgcn_wmma_f32_16x16x32_bf16(
            false, af.v, false, bf.v, (short)0, acc, false, false);
    }

    // store transposed bf16: lane column e = c0+lr, rows n0 + 8*half + j
    US8 pk;
#pragma unroll
    for (int j = 0; j < 8; ++j) pk.s[j] = f2bf(acc[j]);
    size_t tbase = ((size_t)(h * B_ + b) * D_ + (c0 + lr)) * N_ + n0 + half * 8;
    *(uint4*)(WhT + tbase) = pk.u;

    // f1/f2 partial dot over this wave's 16 columns, deterministic reduce
    const float* ab = gata + (size_t)(hop * HEADS_ + h) * 2 * D_;
    const float a1v = ab[c0 + lr];
    const float a2v = ab[D_ + c0 + lr];
    float c1[8], c2[8];
#pragma unroll
    for (int j = 0; j < 8; ++j) { c1[j] = acc[j] * a1v; c2[j] = acc[j] * a2v; }
#pragma unroll
    for (int off = 1; off < 16; off <<= 1) {
#pragma unroll
        for (int j = 0; j < 8; ++j) {
            c1[j] += __shfl_xor(c1[j], off, 32);
            c2[j] += __shfl_xor(c2[j], off, 32);
        }
    }
    if (lr == 0) {
#pragma unroll
        for (int j = 0; j < 8; ++j) {
            pf1[wave][j + 8 * half] = c1[j];
            pf2[wave][j + 8 * half] = c2[j];
        }
    }
    __syncthreads();
    if (tid < 16) {
        float s1 = 0.f, s2 = 0.f;
#pragma unroll
        for (int w = 0; w < 8; ++w) { s1 += pf1[w][tid]; s2 += pf2[w][tid]; }
        size_t o = (size_t)(h * B_ + b) * N_ + n0 + tid;
        f1[o] = s1; f2[o] = s2;
    }
}

// ---------------------------------------------------------------------------
// Fused masked-softmax attention + aggregation, head-mean:
//   out[b,n,:] = (1/H) sum_h softmax_m(mask(lrelu(f1+f2)))[n,m] * Wh[h,b,m,:]
// Block = 128 threads = 4 waves, one wave per head, 16-row tile per block.
// The (head-invariant) adjacency mask is built ONCE per block as a bitmask in
// LDS; softmax passes then run entirely out of LDS/registers. Head-mean is a
// deterministic LDS partial-sum combine (no fp atomics).
// grid: dim3(N/16, B), 128 threads
// ---------------------------------------------------------------------------
__global__ __launch_bounds__(128) void attn_kernel(
    const unsigned short* __restrict__ WhT,   // [H,B,D,N] bf16
    const float* __restrict__ f1, const float* __restrict__ f2,
    const float* __restrict__ adj,
    const int* __restrict__ kb, const int* __restrict__ cv,
    float* __restrict__ outb)                 // [B,N,D]
{
    __shared__ float    sf2[HEADS_][N_];      // 16 KB
    __shared__ unsigned smask[16][N_ / 32];   // 2 KB: bit m of row r
    __shared__ float    spart[HEADS_][16 * D_]; // 32 KB

    const int ntile = blockIdx.x, b = blockIdx.y;
    const int tid = threadIdx.x;
    const int h = tid >> 5;                   // wave == head
    const int lane = tid & 31;
    const int half = lane >> 4, r = lane & 15;
    const int ctx = kb[b] + cv[b];

    // ---- build head-invariant mask bits: 16 rows x 1024 cols, once ----
    {
        int row  = tid >> 3;                  // 0..15
        int nrow = ntile * 16 + row;
        int cbase = (tid & 7) * 128;
        const float* ar = adj + ((size_t)b * N_ + nrow) * N_ + cbase;
        __builtin_prefetch(ar, 0, 0);
#pragma unroll
        for (int w = 0; w < 4; ++w) {
            unsigned wd = 0;
#pragma unroll
            for (int c = 0; c < 32; c += 4) {
                float4 f = *(const float4*)(ar + w * 32 + c);
                if (f.x > 0.f) wd |= 1u << (c + 0);
                if (f.y > 0.f) wd |= 1u << (c + 1);
                if (f.z > 0.f) wd |= 1u << (c + 2);
                if (f.w > 0.f) wd |= 1u << (c + 3);
            }
            int dpos = nrow - cbase - w * 32;   // self-loop for padded slots
            if (nrow >= ctx && dpos >= 0 && dpos < 32) wd |= 1u << dpos;
            smask[row][(cbase >> 5) + w] = wd;
        }
    }
    // ---- stage this head's f2 row into LDS ----
    {
        const float* f2g = f2 + (size_t)(h * B_ + b) * N_;
        for (int t = lane; t < N_; t += 32) sf2[h][t] = f2g[t];
    }
    __syncthreads();

    const float* sf2h = sf2[h];
    const unsigned* smaskRow = smask[r];
    const float f1r = f1[(size_t)(h * B_ + b) * N_ + ntile * 16 + r];
    const int nrow = ntile * 16 + r;
    (void)nrow;

    // ---- pass 1a: row max (lane halves take contiguous half-ranges) ----
    const int mb = half * 512;
    float mx = NEG_INF_;
    for (int wi = 0; wi < 16; ++wi) {
        unsigned wd = smaskRow[(mb >> 5) + wi];
        const float* sp = sf2h + mb + wi * 32;
#pragma unroll
        for (int t = 0; t < 32; t += 4) {
            float4 f = *(const float4*)(sp + t);
            mx = fmaxf(mx, ((wd >> (t + 0)) & 1) ? lrelu(f1r + f.x) : NEG_INF_);
            mx = fmaxf(mx, ((wd >> (t + 1)) & 1) ? lrelu(f1r + f.y) : NEG_INF_);
            mx = fmaxf(mx, ((wd >> (t + 2)) & 1) ? lrelu(f1r + f.z) : NEG_INF_);
            mx = fmaxf(mx, ((wd >> (t + 3)) & 1) ? lrelu(f1r + f.w) : NEG_INF_);
        }
    }
    mx = fmaxf(mx, __shfl_xor(mx, 16, 32));

    // ---- pass 1b: sum of exp ----
    float sm = 0.f;
    for (int wi = 0; wi < 16; ++wi) {
        unsigned wd = smaskRow[(mb >> 5) + wi];
        const float* sp = sf2h + mb + wi * 32;
#pragma unroll
        for (int t = 0; t < 32; t += 4) {
            float4 f = *(const float4*)(sp + t);
            float v0 = ((wd >> (t + 0)) & 1) ? lrelu(f1r + f.x) : NEG_INF_;
            float v1 = ((wd >> (t + 1)) & 1) ? lrelu(f1r + f.y) : NEG_INF_;
            float v2 = ((wd >> (t + 2)) & 1) ? lrelu(f1r + f.z) : NEG_INF_;
            float v3 = ((wd >> (t + 3)) & 1) ? lrelu(f1r + f.w) : NEG_INF_;
            sm += __expf(v0 - mx) + __expf(v1 - mx) +
                  __expf(v2 - mx) + __expf(v3 - mx);
        }
    }
    sm += __shfl_xor(sm, 16, 32);
    const float rinv = 1.f / sm;

    // ---- pass 2: WMMA over m in steps of 32 ----
    v8f acc[8] = {};
    const size_t whbase = (size_t)(h * B_ + b) * D_ * N_;
    for (int kk = 0; kk < N_; kk += 32) {
        unsigned wd = smaskRow[kk >> 5];
        float fv[16];
        *(float4*)&fv[0]  = *(const float4*)&sf2h[kk + half * 8];
        *(float4*)&fv[4]  = *(const float4*)&sf2h[kk + half * 8 + 4];
        *(float4*)&fv[8]  = *(const float4*)&sf2h[kk + half * 8 + 16];
        *(float4*)&fv[12] = *(const float4*)&sf2h[kk + half * 8 + 20];
        BF16x16 af;
#pragma unroll
        for (int i = 0; i < 16; ++i) {
            int bit = ((i >> 3) << 4) + half * 8 + (i & 7);
            float e = lrelu(f1r + fv[i]);
            float v = ((wd >> bit) & 1) ? e : NEG_INF_;
            af.s[i] = f2bf(__expf(v - mx) * rinv);
        }
#pragma unroll
        for (int ct = 0; ct < 8; ++ct) {
            BF16x16 bf;
            const unsigned short* bp =
                WhT + whbase + (size_t)(ct * 16 + r) * N_ + kk + half * 8;
            bf.u[0] = *(const uint4*)bp;
            bf.u[1] = *(const uint4*)(bp + 16);
            acc[ct] = __builtin_amdgcn_wmma_f32_16x16x32_bf16(
                false, af.v, false, bf.v, (short)0, acc[ct], false, false);
        }
    }

    // ---- deterministic head-mean combine via LDS partials ----
#pragma unroll
    for (int ct = 0; ct < 8; ++ct) {
#pragma unroll
        for (int j = 0; j < 8; ++j)
            spart[h][(j + 8 * half) * D_ + ct * 16 + r] = acc[ct][j];
    }
    __syncthreads();
    for (int idx = tid; idx < 16 * D_; idx += 128) {
        float s = spart[0][idx] + spart[1][idx] + spart[2][idx] + spart[3][idx];
        int n = ntile * 16 + (idx >> 7);
        outb[((size_t)b * N_ + n) * D_ + (idx & 127)] = s * (1.f / HEADS_);
    }
}

// ---------------------------------------------------------------------------
// prob_logits[b,n] = dot(A[b,n,:], u[b,:])   (one wave per (b,n))
// grid: B*N/8 blocks, 256 threads
// ---------------------------------------------------------------------------
__global__ __launch_bounds__(256) void logits_kernel(
    const float* __restrict__ A, const float* __restrict__ u,
    float* __restrict__ logits)
{
    int w = blockIdx.x * 8 + (threadIdx.x >> 5);
    int lane = threadIdx.x & 31;
    int b = w >> 10;
    const float* ar = A + (size_t)w * D_;
    const float* ur = u + b * D_;
    float s = 0.f;
#pragma unroll
    for (int t = 0; t < 4; ++t) s += ar[lane + 32 * t] * ur[lane + 32 * t];
#pragma unroll
    for (int off = 16; off; off >>= 1) s += __shfl_xor(s, off, 32);
    if (lane == 0) logits[w] = s;
}

// softmax over n per batch; grid: B blocks, 256 threads
__global__ __launch_bounds__(256) void softmax_kernel(
    const float* __restrict__ logits, float* __restrict__ p)
{
    __shared__ float red[256];
    int b = blockIdx.x, tid = threadIdx.x;
    const float* x = logits + b * N_;
    float mx = -3.4e38f;
    for (int t = tid; t < N_; t += 256) mx = fmaxf(mx, x[t]);
    red[tid] = mx; __syncthreads();
    for (int s = 128; s; s >>= 1) {
        if (tid < s) red[tid] = fmaxf(red[tid], red[tid + s]);
        __syncthreads();
    }
    mx = red[0]; __syncthreads();
    float sm = 0.f;
    for (int t = tid; t < N_; t += 256) sm += __expf(x[t] - mx);
    red[tid] = sm; __syncthreads();
    for (int s = 128; s; s >>= 1) {
        if (tid < s) red[tid] += red[tid + s];
        __syncthreads();
    }
    float inv = 1.f / red[0];
    for (int t = tid; t < N_; t += 256) p[b * N_ + t] = __expf(x[t] - mx) * inv;
}

// u[b,:] += sum_n Cm[b,n,:] * p[b,n]; grid: B blocks, 128 threads
__global__ __launch_bounds__(128) void uupdate_kernel(
    const float* __restrict__ Cm, const float* __restrict__ p,
    float* __restrict__ u)
{
    int b = blockIdx.x, d = threadIdx.x;
    float acc = u[b * D_ + d];
    const float* cb = Cm + (size_t)b * N_ * D_ + d;
    const float* pb = p + b * N_;
    for (int n = 0; n < N_; ++n) acc += cb[(size_t)n * D_] * pb[n];
    u[b * D_ + d] = acc;
}

__global__ void copyf_kernel(const float* __restrict__ src,
                             float* __restrict__ dst, int n)
{
    int i = blockIdx.x * blockDim.x + threadIdx.x;
    if (i < n) dst[i] = src[i];
}

// out = [sigmoid(logits) | u | logits]
__global__ __launch_bounds__(256) void final_kernel(
    const float* __restrict__ logits, const float* __restrict__ u,
    float* __restrict__ out)
{
    int i = blockIdx.x * 256 + threadIdx.x;
    if (i < B_ * N_) {
        float x = logits[i];
        out[i] = 1.f / (1.f + __expf(-x));
        out[B_ * N_ + B_ * D_ + i] = x;
    }
    if (i < B_ * D_) out[B_ * N_ + i] = u[i];
}

// ---------------------------------------------------------------------------
extern "C" void kernel_launch(void* const* d_in, const int* in_sizes, int n_in,
                              void* d_out, int out_size, void* d_ws,
                              size_t ws_size, hipStream_t stream)
{
    (void)in_sizes; (void)n_in; (void)out_size; (void)ws_size;
    const int*   story  = (const int*)d_in[0];
    const int*   kb     = (const int*)d_in[1];
    const int*   cv     = (const int*)d_in[2];
    const float* hidden = (const float*)d_in[3];
    const float* dh     = (const float*)d_in[4];
    const float* adj    = (const float*)d_in[5];
    const float* emb    = (const float*)d_in[6];
    const float* gatW   = (const float*)d_in[7];  // [1,HOPS+1,H,D,D]
    const float* gata   = (const float*)d_in[8];  // [1,HOPS+1,H,2D]

    char* base = (char*)d_ws;
    size_t off = 0;
    auto take = [&](size_t bytes) -> void* {
        void* p = base + off;
        off += (bytes + 255) & ~(size_t)255;
        return p;
    };
    unsigned short* Xbf  = (unsigned short*)take((size_t)B_ * N_ * D_ * 2);
    unsigned short* WhT  = (unsigned short*)take((size_t)HEADS_ * B_ * D_ * N_ * 2);
    float* f1     = (float*)take((size_t)HEADS_ * B_ * N_ * 4);
    float* f2     = (float*)take((size_t)HEADS_ * B_ * N_ * 4);
    float* bufA   = (float*)take((size_t)B_ * N_ * D_ * 4);
    float* bufC   = (float*)take((size_t)B_ * N_ * D_ * 4);
    float* logits = (float*)take((size_t)B_ * N_ * 4);
    float* psoft  = (float*)take((size_t)B_ * N_ * 4);
    float* u      = (float*)take((size_t)B_ * D_ * 4);

    copyf_kernel<<<(B_ * D_ + 255) / 256, 256, 0, stream>>>(hidden, u, B_ * D_);

    auto gat_pass = [&](int hop_idx, float* dst) {
        embed_kernel<<<B_ * N_, 128, 0, stream>>>(story, kb, cv, dh, emb,
                                                  hop_idx, Xbf);
        wh_gemm_kernel<<<dim3(B_ * N_ / 16, HEADS_), 256, 0, stream>>>(
            Xbf, gatW, gata, hop_idx, WhT, f1, f2);
        attn_kernel<<<dim3(N_ / 16, B_), 128, 0, stream>>>(WhT, f1, f2, adj,
                                                           kb, cv, dst);
    };

    for (int hop = 0; hop < HOPS_; ++hop) {
        gat_pass(hop, bufA);
        logits_kernel<<<B_ * N_ / 8, 256, 0, stream>>>(bufA, u, logits);
        softmax_kernel<<<B_, 256, 0, stream>>>(logits, psoft);
        gat_pass(hop + 1, bufC);
        uupdate_kernel<<<B_, 128, 0, stream>>>(bufC, psoft, u);
    }
    final_kernel<<<(B_ * N_ + 255) / 256, 256, 0, stream>>>(logits, u,
                                                            (float*)d_out);
}
